// modelLSTMVent_77721728188836
// MI455X (gfx1250) — matrix operand order repaired and checked by reading, not compiled, not run on hardware
//
#include <hip/hip_runtime.h>

typedef __attribute__((ext_vector_type(16))) _Float16 v16h;
typedef __attribute__((ext_vector_type(8)))  float    v8f;

namespace {
constexpr int kH = 32;     // hidden
constexpr int kT = 129;    // seq len
constexpr int kS = 4;      // streams
constexpr int kG = 4 * kH; // 128 gates
}

// Intra-wave LDS ordering: CDNA5 DS ops from one wave execute in order at the
// hardware level; we only need to stop the compiler from reordering them.
__device__ __forceinline__ void lds_fence() {
  asm volatile("" ::: "memory");
  __builtin_amdgcn_wave_barrier();
  asm volatile("" ::: "memory");
}

__device__ __forceinline__ float fexp(float x) {
  return __builtin_amdgcn_exp2f(x * 1.4426950408889634f);
}
__device__ __forceinline__ float fsigm(float x) {
  return __builtin_amdgcn_rcpf(1.0f + fexp(-x));
}
__device__ __forceinline__ float ftanh(float x) {
  float e = fexp(2.0f * x);
  return (e - 1.0f) * __builtin_amdgcn_rcpf(e + 1.0f);
}

__device__ __forceinline__ v8f wmma_f16f32(v16h a, v16h b, v8f c) {
  // D = A(16x32 f16) * B(32x16 f16) + C(f32)
  return __builtin_amdgcn_wmma_f32_16x16x32_f16(false, a, false, b, (short)0, c,
                                                false, false);
}

// Build the 16x32 f16 A-fragment for one row tile of a row-major 128x32 f32
// matrix. Lane m=lane&15 holds row (tile*16+m); half j maps to
// K = j + (lane>=16 ? 16 : 0)  -> one contiguous 16-float run per lane.
__device__ __forceinline__ v16h load_afrag(const float* __restrict__ w,
                                           int tile, int m, int kb) {
  const float* rp = w + (tile * 16 + m) * kH;
  v16h a;
#pragma unroll
  for (int j = 0; j < 8; ++j) a[j] = (_Float16)rp[kb + j];
#pragma unroll
  for (int j = 0; j < 8; ++j) a[8 + j] = (_Float16)rp[kb + 8 + j];
  return a;
}

__global__ __launch_bounds__(128, 1) void lstm4_wmma_kernel(
    const float* __restrict__ x, const float* __restrict__ Wih0,
    const float* __restrict__ Whh0, const float* __restrict__ bih0,
    const float* __restrict__ bhh0, const float* __restrict__ Wih1,
    const float* __restrict__ Whh1, const float* __restrict__ bih1,
    const float* __restrict__ bhh1, const float* __restrict__ Wt,
    const float* __restrict__ bt, const float* __restrict__ Wf1,
    const float* __restrict__ bf1, const float* __restrict__ Wf2,
    const float* __restrict__ bf2, float* __restrict__ out) {
  __shared__ alignas(32) _Float16 hbuf[kS][64];  // B-fragment staging (h0|h1)
  __shared__ alignas(16) float gbuf[kS][kG];     // 128 gate pre-activations
  __shared__ float xbuf[kS][kT];
  __shared__ float hlast[kS][kH];

  const int tid  = (int)threadIdx.x;
  const int s    = tid >> 5;            // wave = stream
  const int l    = tid & 31;            // lane = hidden index
  const int m    = l & 15;
  const int kb   = (l & 16) ? 16 : 0;   // A-fragment K base per lane half
  const int rh   = (l & 16) ? 8 : 0;    // C/D row half (M = r + 8*(lane>=16))
  const int boff = (l & 16) ? 16 : 0;   // B-fragment K base per lane half

  // Stage x into LDS once (516 floats).
  for (int i = tid; i < kS * kT; i += 128) xbuf[i / kT][i % kT] = x[i];

  // Convert this stream's recurrent weights to f16 A-fragments, kept in VGPRs.
  const float* whh0 = Whh0 + s * kG * kH;
  const float* wih1 = Wih1 + s * kG * kH;
  const float* whh1 = Whh1 + s * kG * kH;
  v16h aWhh0[8], aWih1[8], aWhh1[8];
#pragma unroll
  for (int t8 = 0; t8 < 8; ++t8) {
    aWhh0[t8] = load_afrag(whh0, t8, m, kb);
    aWih1[t8] = load_afrag(wih1, t8, m, kb);
    aWhh1[t8] = load_afrag(whh1, t8, m, kb);
  }

  // Per-lane (gate = l, l+32, l+64, l+96) input weights and fused biases.
  const int gbase = s * kG;
  const float w0i = Wih0[gbase + l];
  const float w0f = Wih0[gbase + kH + l];
  const float w0g = Wih0[gbase + 2 * kH + l];
  const float w0o = Wih0[gbase + 3 * kH + l];
  const float b0i = bih0[gbase + l] + bhh0[gbase + l];
  const float b0f = bih0[gbase + kH + l] + bhh0[gbase + kH + l];
  const float b0g = bih0[gbase + 2 * kH + l] + bhh0[gbase + 2 * kH + l];
  const float b0o = bih0[gbase + 3 * kH + l] + bhh0[gbase + 3 * kH + l];
  const float b1i = bih1[gbase + l] + bhh1[gbase + l];
  const float b1f = bih1[gbase + kH + l] + bhh1[gbase + kH + l];
  const float b1g = bih1[gbase + 2 * kH + l] + bhh1[gbase + 2 * kH + l];
  const float b1o = bih1[gbase + 3 * kH + l] + bhh1[gbase + 3 * kH + l];

  __syncthreads();  // xbuf ready

  _Float16* hb = &hbuf[s][0];
  float*    gb = &gbuf[s][0];

  float h0 = 0.f, c0 = 0.f, h1 = 0.f, c1 = 0.f;

#pragma unroll 1
  for (int t = 0; t < kT; ++t) {
    const float xt = xbuf[s][t];

    // -------- layer 0: g = Whh0*h0 (+ Wih0*x + bias folded per-lane) --------
    hb[l] = (_Float16)h0;
    lds_fence();
    const v16h bh = *(const v16h*)(hb + boff);  // h0 replicated in all N cols
#pragma unroll
    for (int grp = 0; grp < 2; ++grp) {
      const int t0 = grp * 4;
      // Distinct accumulators -> 4 independent, overlappable WMMAs.
      v8f a0 = {}, a1 = {}, a2 = {}, a3 = {};
      a0 = wmma_f16f32(aWhh0[t0 + 0], bh, a0);
      a1 = wmma_f16f32(aWhh0[t0 + 1], bh, a1);
      a2 = wmma_f16f32(aWhh0[t0 + 2], bh, a2);
      a3 = wmma_f16f32(aWhh0[t0 + 3], bh, a3);
      if (m == 0) {  // all D columns equal -> lanes 0 and 16 scatter rows
#pragma unroll
        for (int r = 0; r < 8; ++r) {
          gb[(t0 + 0) * 16 + rh + r] = a0[r];
          gb[(t0 + 1) * 16 + rh + r] = a1[r];
          gb[(t0 + 2) * 16 + rh + r] = a2[r];
          gb[(t0 + 3) * 16 + rh + r] = a3[r];
        }
      }
    }
    lds_fence();
    {
      const float gi = gb[l]          + w0i * xt + b0i;
      const float gf = gb[kH + l]     + w0f * xt + b0f;
      const float gg = gb[2 * kH + l] + w0g * xt + b0g;
      const float go = gb[3 * kH + l] + w0o * xt + b0o;
      c0 = fsigm(gf) * c0 + fsigm(gi) * ftanh(gg);
      h0 = fsigm(go) * ftanh(c0);
    }

    // -------- layer 1: g = Wih1*h0 + Whh1*h1 (K=64 via chained WMMA) --------
    hb[l]      = (_Float16)h0;
    hb[32 + l] = (_Float16)h1;
    lds_fence();
    const v16h bh0 = *(const v16h*)(hb + boff);
    const v16h bh1 = *(const v16h*)(hb + 32 + boff);
#pragma unroll
    for (int grp = 0; grp < 2; ++grp) {
      const int t0 = grp * 4;
      v8f a0 = {}, a1 = {}, a2 = {}, a3 = {};
      a0 = wmma_f16f32(aWih1[t0 + 0], bh0, a0);
      a1 = wmma_f16f32(aWih1[t0 + 1], bh0, a1);
      a2 = wmma_f16f32(aWih1[t0 + 2], bh0, a2);
      a3 = wmma_f16f32(aWih1[t0 + 3], bh0, a3);
      a0 = wmma_f16f32(aWhh1[t0 + 0], bh1, a0);
      a1 = wmma_f16f32(aWhh1[t0 + 1], bh1, a1);
      a2 = wmma_f16f32(aWhh1[t0 + 2], bh1, a2);
      a3 = wmma_f16f32(aWhh1[t0 + 3], bh1, a3);
      if (m == 0) {
#pragma unroll
        for (int r = 0; r < 8; ++r) {
          gb[(t0 + 0) * 16 + rh + r] = a0[r];
          gb[(t0 + 1) * 16 + rh + r] = a1[r];
          gb[(t0 + 2) * 16 + rh + r] = a2[r];
          gb[(t0 + 3) * 16 + rh + r] = a3[r];
        }
      }
    }
    lds_fence();
    {
      const float gi = gb[l]          + b1i;
      const float gf = gb[kH + l]     + b1f;
      const float gg = gb[2 * kH + l] + b1g;
      const float go = gb[3 * kH + l] + b1o;
      c1 = fsigm(gf) * c1 + fsigm(gi) * ftanh(gg);
      h1 = fsigm(go) * ftanh(c1);
    }
  }

  hlast[s][l] = h1;
  __syncthreads();

  // Tiny output heads (4 scalars) — serial on one lane, negligible.
  if (tid == 0) {
    float acc = bt[0];
    for (int k = 0; k < kS * kH; ++k) acc += hlast[k >> 5][k & 31] * Wt[k];
    out[0] = acc;
    float a1 = bf1[0];
    for (int k = 0; k < kH; ++k) a1 += hlast[1][k] * Wf1[k];
    out[1] = a1;
    float a2 = bf2[0];
    for (int k = 0; k < kH; ++k) a2 += hlast[2][k] * Wf2[k];
    out[2] = a2;
    float a3 = bf2[0];  // reference bug preserved: f3 head reuses Wf2/bf2
    for (int k = 0; k < kH; ++k) a3 += hlast[3][k] * Wf2[k];
    out[3] = a3;
  }
}

extern "C" void kernel_launch(void* const* d_in, const int* in_sizes, int n_in,
                              void* d_out, int out_size, void* d_ws,
                              size_t ws_size, hipStream_t stream) {
  (void)in_sizes; (void)n_in; (void)out_size; (void)d_ws; (void)ws_size;
  lstm4_wmma_kernel<<<dim3(1), dim3(128), 0, stream>>>(
      (const float*)d_in[0],  (const float*)d_in[1],  (const float*)d_in[2],
      (const float*)d_in[3],  (const float*)d_in[4],  (const float*)d_in[5],
      (const float*)d_in[6],  (const float*)d_in[7],  (const float*)d_in[8],
      (const float*)d_in[9],  (const float*)d_in[10], (const float*)d_in[11],
      (const float*)d_in[12], (const float*)d_in[13], (const float*)d_in[14],
      (float*)d_out);
}
